// Head1_47966194762087
// MI455X (gfx1250) — compile-verified, matrix operand order checked by reading, not run on hardware
//
#include <hip/hip_runtime.h>
#include <hip/hip_bf16.h>

// ---------- types for WMMA ----------
typedef __attribute__((ext_vector_type(16))) __bf16 v16bf;
typedef __attribute__((ext_vector_type(8)))  float  v8f;
typedef __attribute__((ext_vector_type(4)))  unsigned int uint4v;

// ---------- constants ----------
#define BATCH 16
#define S 64
#define HW 4096            // S*S
#define MTOT 65536         // BATCH*HW
#define CIN 256
#define CINX 258           // +2 grid channels
#define CIP 288            // padded to multiple of 32
#define CO 512
#define BM 128
#define BN 64
#define LDA 40             // ushorts per LDS A row (80B, padded from 64B)
#define LDB 40
#define KSTEPS 81          // 9 taps * 9 chunks of K=32

__device__ __forceinline__ unsigned short f2bf(float f) {
  union { float f; unsigned int u; } x; x.f = f;
  unsigned int u = x.u;
  unsigned int r = u + 0x7fffu + ((u >> 16) & 1u);
  if ((u & 0x7f800000u) == 0x7f800000u) r = u;   // inf/nan passthrough
  return (unsigned short)(r >> 16);
}

__device__ __forceinline__ float sigmoidf_(float x) { return 1.0f / (1.0f + expf(-x)); }

// LDS byte offset of a generic pointer to a __shared__ object (low 32 bits of
// the flat address inside the LDS aperture are the workgroup-relative offset).
__device__ __forceinline__ unsigned ldsoff(const void* p) {
  return (unsigned)(unsigned long long)p;
}

// 16B global -> LDS async copy (ASYNCcnt-tracked DMA, no VGPR round trip)
__device__ __forceinline__ void async_cp16(unsigned lds_addr, const void* gptr) {
  unsigned long long ga = (unsigned long long)gptr;
  asm volatile("global_load_async_to_lds_b128 %0, %1, off"
               :: "v"(lds_addr), "v"(ga) : "memory");
}

// =====================================================================
// prep_x: NCHW f32 -> [b*4096+h*64+w][288] bf16 (c=256:gx=w, c=257:gy=h, pad 0)
// one block per (b,h): LDS transpose tile 64(w) x 292(c)
// =====================================================================
__global__ __launch_bounds__(256) void prep_x(const float* __restrict__ in,
                                              unsigned short* __restrict__ xbf) {
  __shared__ unsigned short t[64 * 292];
  const int b = blockIdx.x >> 6;
  const int h = blockIdx.x & 63;
  const int tid = threadIdx.x;
  const int wq = tid >> 6;      // 0..3
  const int wcol = tid & 63;    // w index for read phase
  for (int i = 0; i < 64; ++i) {
    int c = i * 4 + wq;
    float v = in[(((size_t)b * CIN + c) << 12) + (h << 6) + wcol];
    t[wcol * 292 + c] = f2bf(v);
  }
  {
    int c = 256 + wq;
    unsigned short val = 0;
    if (c == 256) val = f2bf((float)wcol);
    else if (c == 257) val = f2bf((float)h);
    t[wcol * 292 + c] = val;
  }
  for (int j = tid; j < 64 * 28; j += 256) {
    int w = j / 28, c = 260 + (j % 28);
    t[w * 292 + c] = 0;
  }
  __syncthreads();
  unsigned int* dst = (unsigned int*)xbf;
  for (int j = tid; j < 64 * 144; j += 256) {
    int w = j / 144, cp = j % 144;
    unsigned int lo = t[w * 292 + cp * 2];
    unsigned int hi = t[w * 292 + cp * 2 + 1];
    dst[(((size_t)b << 12) + (h << 6) + w) * 144 + cp] = lo | (hi << 16);
  }
}

// =====================================================================
// prep_w: (co,ci,ky,kx) f32 -> [tap][co][ci_pad 288] bf16
// =====================================================================
__global__ __launch_bounds__(256) void prep_w(const float* __restrict__ wc,
                                              unsigned short* __restrict__ wB) {
  int idx = blockIdx.x * 256 + threadIdx.x;
  const int total = 9 * CO * CIP;
  if (idx >= total) return;
  int c = idx % CIP;
  int t = idx / CIP;
  int co = t % CO;
  int tap = t / CO;
  float v = 0.0f;
  if (c < CINX) v = wc[((size_t)co * CINX + c) * 9 + tap];
  wB[idx] = f2bf(v);
}

// =====================================================================
// conv_wmma: implicit GEMM, D[m][co] = sum_k A[m][k]*B[co][k]
// block tile 128x64, 8 waves as 4(M)x2(N), each wave 2x2 frags of 16x16.
// Double-buffered LDS pipeline fed by global_load_async_to_lds_b128:
// exactly 3 async ops per wave per K-step -> s_wait_asynccnt 0x3 keeps the
// next step's copies in flight while the current step runs on the WMMAs.
// Fused epilogue: store y fp32 + atomic per-channel sum / sumsq for BN.
// =====================================================================
__global__ __launch_bounds__(256) void conv_wmma(const unsigned short* __restrict__ xbf,
                                                 const unsigned short* __restrict__ wB,
                                                 const unsigned short* __restrict__ zpad,
                                                 float* __restrict__ y,
                                                 float* __restrict__ gsum,
                                                 float* __restrict__ gsumsq) {
  __shared__ alignas(16) unsigned short sA[2][BM * LDA];
  __shared__ alignas(16) unsigned short sB[2][BN * LDB];

  const int tid  = threadIdx.x;
  const int lane = tid & 31;
  const int wv   = tid >> 5;
  const int wm   = wv & 3;       // 0..3  (M group of 32 rows)
  const int wn   = wv >> 2;      // 0..1  (N group of 32 cols)
  const int m0   = blockIdx.x * BM;
  const int n0   = blockIdx.y * BN;

  v8f acc[2][2];
#pragma unroll
  for (int i = 0; i < 2; ++i)
#pragma unroll
    for (int j = 0; j < 2; ++j)
#pragma unroll
      for (int e = 0; e < 8; ++e) acc[i][j][e] = 0.0f;

  // A staging mapping: thread -> (row 0..127, half 0..1) ; 16 bf16 per thread
  const int arow = tid >> 1;
  const int ahalf = tid & 1;
  const int am = m0 + arow;
  const int ab = am >> 12;
  const int ahw = am & 4095;
  const int ah = ahw >> 6, aw = ahw & 63;
  // B staging mapping: thread -> (row 0..63, quarter 0..3) ; 8 bf16 per thread
  const int brow = tid >> 2;
  const int bq = tid & 3;
  const int bco = n0 + brow;

  const unsigned ldsA0 = ldsoff(&sA[0][arow * LDA + ahalf * 16]);
  const unsigned ldsA1 = ldsoff(&sA[1][arow * LDA + ahalf * 16]);
  const unsigned ldsB0 = ldsoff(&sB[0][brow * LDB + bq * 8]);
  const unsigned ldsB1 = ldsoff(&sB[1][brow * LDB + bq * 8]);

  // issue one K-step's copies into buffer `buf` (always 3 async ops / wave)
  auto issue = [&](int step, int buf) {
    const int tap = step / 9;
    const int kc  = step - tap * 9;
    const int ky = tap / 3 - 1;
    const int kx = tap - (tap / 3) * 3 - 1;
    const int ayy = ah + ky, axx = aw + kx;
    const bool ain = (ayy >= 0) && (ayy < S) && (axx >= 0) && (axx < S);
    const unsigned short* srcA =
        ain ? (xbf + ((size_t)((ab << 12) + (ayy << 6) + axx)) * CIP +
               kc * 32 + ahalf * 16)
            : zpad;                               // zero pad: halo reads 0
    const unsigned short* srcB =
        wB + ((size_t)(tap * CO + bco)) * CIP + kc * 32 + bq * 8;
    const unsigned la = buf ? ldsA1 : ldsA0;
    const unsigned lb = buf ? ldsB1 : ldsB0;
    async_cp16(la, srcA);
    async_cp16(la + 16u, srcA + 8);
    async_cp16(lb, srcB);
  };

  issue(0, 0);
  issue(1, 1);

  const int k0 = (lane >> 4) << 3;   // 0 or 8 : per-lane K base of fragments
  const int lm = lane & 15;

#pragma unroll 1
  for (int step = 0; step < KSTEPS; ++step) {
    const int cur = step & 1;
    if (step < KSTEPS - 1)
      asm volatile("s_wait_asynccnt 0x3" ::: "memory");  // current buffer done
    else
      asm volatile("s_wait_asynccnt 0x0" ::: "memory");
    __syncthreads();

    const unsigned short* bA = sA[cur];
    const unsigned short* bB = sB[cur];
    v16bf afr[2], bfr[2];
#pragma unroll
    for (int mf = 0; mf < 2; ++mf) {
      const unsigned short* pa = bA + (wm * 32 + mf * 16 + lm) * LDA + k0;
      union { uint4v q[2]; v16bf v; } u;
      u.q[0] = *(const uint4v*)pa;
      u.q[1] = *(const uint4v*)(pa + 16);
      afr[mf] = u.v;
    }
#pragma unroll
    for (int nf = 0; nf < 2; ++nf) {
      const unsigned short* pb = bB + (wn * 32 + nf * 16 + lm) * LDB + k0;
      union { uint4v q[2]; v16bf v; } u;
      u.q[0] = *(const uint4v*)pb;
      u.q[1] = *(const uint4v*)(pb + 16);
      bfr[nf] = u.v;
    }
#pragma unroll
    for (int mf = 0; mf < 2; ++mf)
#pragma unroll
      for (int nf = 0; nf < 2; ++nf)
        acc[mf][nf] = __builtin_amdgcn_wmma_f32_16x16x32_bf16(
            false, afr[mf], false, bfr[nf], (short)0, acc[mf][nf], false, false);

    __syncthreads();                       // everyone done reading buf `cur`
    if (step + 2 < KSTEPS) issue(step + 2, cur);
  }

  // ---- epilogue: store y[m][co] fp32, accumulate BN stats ----
  const int rowoff = (lane >> 4) << 3;  // C layout: lanes>=16 hold rows M+8..M+15
#pragma unroll
  for (int nf = 0; nf < 2; ++nf) {
    float csum = 0.0f, csq = 0.0f;
    const int co = n0 + wn * 32 + nf * 16 + lm;
#pragma unroll
    for (int mf = 0; mf < 2; ++mf) {
      const int mbase = m0 + wm * 32 + mf * 16 + rowoff;
#pragma unroll
      for (int e = 0; e < 8; ++e) {
        float v = acc[mf][nf][e];
        y[(size_t)(mbase + e) * CO + co] = v;
        csum += v; csq += v * v;
      }
    }
    csum += __shfl_xor(csum, 16, 32);   // partner lane holds same column co
    csq  += __shfl_xor(csq, 16, 32);
    if (lane < 16) {
      atomicAdd(&gsum[co], csum);
      atomicAdd(&gsumsq[co], csq);
    }
  }
}

// =====================================================================
// bn_finalize: per-channel scale/shift from batch stats
// =====================================================================
__global__ void bn_finalize(const float* __restrict__ gsum, const float* __restrict__ gsumsq,
                            const float* __restrict__ gamma, const float* __restrict__ beta,
                            float* __restrict__ scl, float* __restrict__ shf) {
  int c = blockIdx.x * blockDim.x + threadIdx.x;
  if (c < CO) {
    const float inv = 1.0f / (float)MTOT;
    float m = gsum[c] * inv;
    float var = gsumsq[c] * inv - m * m;
    float s = gamma[c] * rsqrtf(var + 1e-5f);
    scl[c] = s;
    shf[c] = beta[c] - m * s;
  }
}

// =====================================================================
// decode: one wave per pixel. 58 dot-products over 512 channels with BN+leaky
// applied on the fly; butterfly reduce; lane-parallel decode + writes.
// =====================================================================
__global__ __launch_bounds__(256) void decode_kernel(
    const float* __restrict__ y, const float* __restrict__ scl, const float* __restrict__ shf,
    const float* __restrict__ wloc, const float* __restrict__ bloc,
    const float* __restrict__ wmask, const float* __restrict__ bmask,
    float* __restrict__ out) {
  __shared__ float scm[8][64];
  const int lane = threadIdx.x & 31;
  const int wv = threadIdx.x >> 5;
  const int pos = blockIdx.x * 8 + wv;
  const int b = pos >> 12;
  const int hw = pos & 4095;
  const int h = hw >> 6, w = hw & 63;

  float acc[58];
#pragma unroll
  for (int j = 0; j < 58; ++j) acc[j] = 0.0f;

  const float* yrow = y + (size_t)pos * CO;
  for (int i = 0; i < 16; ++i) {
    const int co = i * 32 + lane;
    float v = yrow[co] * scl[co] + shf[co];
    v = v > 0.0f ? v : 0.1f * v;               // leaky relu
#pragma unroll
    for (int j = 0; j < 10; ++j) acc[j] += v * wloc[j * CO + co];
#pragma unroll
    for (int j = 0; j < 48; ++j) acc[10 + j] += v * wmask[j * CO + co];
  }
#pragma unroll
  for (int j = 0; j < 58; ++j) {
    float r = acc[j];
    r += __shfl_xor(r, 1, 32); r += __shfl_xor(r, 2, 32);
    r += __shfl_xor(r, 4, 32); r += __shfl_xor(r, 8, 32);
    r += __shfl_xor(r, 16, 32);
    acc[j] = r;
  }
  float p[10], cm[48];
#pragma unroll
  for (int j = 0; j < 10; ++j) p[j] = acc[j] + bloc[j];
#pragma unroll
  for (int j = 0; j < 48; ++j) cm[j] = acc[10 + j] + bmask[j];

  if (lane == 0) {
#pragma unroll
    for (int j = 0; j < 10; ++j) scm[wv][j] = p[j];
#pragma unroll
    for (int j = 0; j < 48; ++j) scm[wv][10 + j] = cm[j];
  }
  __syncthreads();

  // ---------------- decode math (uniform per wave) ----------------
  const float st = 8.0f, inv_st = 0.125f;
  const float gX = (float)w, gY = (float)h;
  const float l0 = p[0] * p[0] * st, l1 = p[1] * p[1] * st;
  const float l2 = p[2] * p[2] * st, l3 = p[3] * p[3] * st;
  const float cx = gX * st + 4.0f, cy = gY * st + 4.0f;
  const float xmin = cx - l3, ymin = cy - l0, xmax = cx + l1, ymax = cy + l2;
  const float wd = l1 + l3, ht = l0 + l2;
  const float xc = 0.5f * (xmax + xmin), yc = 0.5f * (ymax + ymin);
  const float r = sigmoidf_(p[8]);
  const float maskr = (r > 0.9f) ? 0.0f : 1.0f;
  const float s0 = sigmoidf_(p[4]) * maskr, s1 = sigmoidf_(p[5]) * maskr;
  const float s2 = sigmoidf_(p[6]) * maskr, s3 = sigmoidf_(p[7]) * maskr;
  const float conf = sigmoidf_(p[9]);
  const float x1 = xmin + s0 * wd, x7 = xmax - s2 * wd;
  const float y5 = ymin + s1 * ht, y3 = ymax - s3 * ht;
  const float x_obb = 0.5f * (x1 + x7), y_obb = 0.5f * (y5 + y3);
  const float off0 = sigmoidf_(cm[18]), off1 = sigmoidf_(cm[19]);
  const float off2 = sigmoidf_(cm[20]), off3 = sigmoidf_(cm[21]);
  const float eps = ceilf(0.01f * wd);

  const bool c0c = x1 < xmin + eps;
  const float xp0 = c0c ? x1 : xmin + off0 * (x1 - xmin);
  const float yp0 = c0c ? (ymin + off0 * (y3 - ymin))
                        : (y3 - ymin) / (xmin - x1 + 1e-8f) * (xp0 - x1) + ymin;
  const bool c2c = x1 > xmax - eps;
  const float xp2 = c2c ? x1 : xmax - off1 * (xmax - x1);
  const float yp2 = c2c ? (ymin + off1 * (y5 - ymin))
                        : (y5 - ymin) / (xmax - x1 + 1e-8f) * (xp2 - x1) + ymin;
  const bool c6c = x7 < xmin + eps;
  const float xp6 = c6c ? x7 : xmin + off2 * (x7 - xmin);
  const float yp6 = c6c ? (ymax - off2 * (ymax - y3))
                        : (y3 - ymax) / (xmin - x7 + 1e-8f) * (xp6 - x7) + ymax;
  const bool c8c = x7 > xmax - eps;
  const float xp8 = c8c ? x7 : xmax - off3 * (xmax - x7);
  const float yp8 = c8c ? (ymax - off3 * (ymax - y5))
                        : (y5 - ymax) / (xmax - x7 + 1e-8f) * (xp8 - x7) + ymax;

  const float width = 0.5f * (sqrtf((x1 - xmax) * (x1 - xmax) + (ymin - y5) * (ymin - y5)) +
                              sqrtf((xmin - x7) * (xmin - x7) + (y3 - ymax) * (y3 - ymax)));
  const float height = 0.5f * (sqrtf((xmin - x1) * (xmin - x1) + (y3 - ymin) * (y3 - ymin)) +
                               sqrtf((x7 - xmax) * (x7 - xmax) + (ymax - y5) * (ymax - y5)));
  const float angle = 0.5f * (atanf((y5 - ymin) / (xmax - x1 + 1e-4f)) +
                              atanf((ymax - y3) / (x7 - xmin + 1e-4f)));
  const float ca = cosf(angle), sa = sinf(angle);

  float m8 = cm[40];
#pragma unroll
  for (int j = 41; j < 48; ++j) m8 = fmaxf(m8, cm[j]);
  float ssum = 0.0f;
#pragma unroll
  for (int j = 40; j < 48; ++j) ssum += expf(cm[j] - m8);
  const float rsum = 1.0f / ssum;

  const size_t O0 = 0;          // p           (B,S,S,10)
  const size_t O1 = 655360;     // bbox        (B,S,S,14)
  const size_t O2 = 1572864;    // offsets_loc (B,18,S,S)
  const size_t O3 = 2752512;    // offsets_cls (B,18,S,S)
  const size_t O4 = 3932160;    // mask_loc    (B,9,S,S)
  const size_t O5 = 4521984;    // mask_cls    (B,9,S,S)
  const size_t O6 = 5111808;    // weight_c8   (B,8,S,S)
  const size_t O7 = 5636096;    // offsets_d   (B*S*S,18)

  if (lane < 10) out[O0 + (size_t)pos * 10 + lane] = p[lane];

  if (lane < 14) {
    float bb;
    switch (lane) {
      case 0: bb = xc; break;   case 1: bb = yc; break;
      case 2: bb = wd; break;   case 3: bb = ht; break;
      case 4: bb = s0; break;   case 5: bb = s1; break;
      case 6: bb = s2; break;   case 7: bb = s3; break;
      case 8: bb = r;  break;   case 9: bb = l0; break;
      case 10: bb = l1; break;  case 11: bb = l2; break;
      case 12: bb = l3; break;  default: bb = conf; break;
    }
    out[O1 + (size_t)pos * 14 + lane] = bb;
  }

  if (lane < 18) {
    float v;
    switch (lane) {
      case 0:  v = yp0 * inv_st + 1.0f - gY; break;
      case 1:  v = xp0 * inv_st + 1.0f - gX; break;
      case 2:  v = ymin * inv_st + 1.0f - gY; break;
      case 3:  v = x1 * inv_st - gX; break;
      case 4:  v = yp2 * inv_st + 1.0f - gY; break;
      case 5:  v = xp2 * inv_st - 1.0f - gX; break;
      case 6:  v = y3 * inv_st - gY; break;
      case 7:  v = xmin * inv_st + 1.0f - gX; break;
      case 8:  v = 0.0f; break;
      case 9:  v = 0.0f; break;
      case 10: v = y5 * inv_st - gY; break;
      case 11: v = xmax * inv_st - 1.0f - gX; break;
      case 12: v = yp6 * inv_st - 1.0f - gY; break;
      case 13: v = xp6 * inv_st + 1.0f - gX; break;
      case 14: v = ymax * inv_st - 1.0f - gY; break;
      case 15: v = x7 * inv_st - gX; break;
      case 16: v = yp8 * inv_st - 1.0f - gY; break;
      default: v = xp8 * inv_st - 1.0f - gX; break;
    }
    out[O2 + (((size_t)b * 18 + lane) << 12) + hw] = v;
  }

  if (lane < 9) {
    out[O4 + (((size_t)b * 9 + lane) << 12) + hw] = sigmoidf_(scm[wv][10 + lane]);
    out[O5 + (((size_t)b * 9 + lane) << 12) + hw] = sigmoidf_(scm[wv][19 + lane]);
  }
  if (lane < 8) {
    out[O6 + (((size_t)b * 8 + lane) << 12) + hw] = expf(scm[wv][50 + lane] - m8) * rsum;
  }

  if (lane < 9) {
    const int j = lane;
    const int jr = j / 3, jc = j % 3;
    const float D = 0.70710678118654752440f;
    const float dyv = (float)(1 - jr) * ((jc == 1) ? 1.0f : D);
    const float dxv = (float)(1 - jc) * ((jr == 1) ? 1.0f : D);
    const float dcy = sigmoidf_(scm[wv][32 + j]);   // dc_off[0:9]
    const float dcx = sigmoidf_(scm[wv][41 + j]);   // dc_off[9:18]
    const float xd = x_obb - 0.5f * width + dcx * width;
    const float yd = y_obb - 0.5f * height + dcy * height;
    const float xd0 = ca * (xd - x_obb) - sa * (yd - y_obb) + x_obb;
    const float xd_ = (fmaxf(xd0, xmin) <= xmax) ? xd0 : xmax;
    const float yd0 = sa * (xd - x_obb) + ca * (yd - y_obb) + y_obb;
    const float yd_ = (fmaxf(yd0, ymin) <= ymax) ? yd0 : ymax;
    const float dc_y = yd_ * inv_st + dyv - gY;
    const float dc_x = xd_ * inv_st + dxv - gX;
    out[O3 + (((size_t)b * 18 + 2 * j) << 12) + hw] = dc_y;
    out[O3 + (((size_t)b * 18 + 2 * j + 1) << 12) + hw] = dc_x;
    out[O7 + (size_t)pos * 18 + j] = xd_;
    out[O7 + (size_t)pos * 18 + 9 + j] = yd_;
  }
}

// =====================================================================
// launcher
// =====================================================================
extern "C" void kernel_launch(void* const* d_in, const int* in_sizes, int n_in,
                              void* d_out, int out_size, void* d_ws, size_t ws_size,
                              hipStream_t stream) {
  const float* input1 = (const float*)d_in[0];
  const float* w_conv = (const float*)d_in[1];
  const float* gamma  = (const float*)d_in[2];
  const float* beta   = (const float*)d_in[3];
  const float* w_loc  = (const float*)d_in[4];
  const float* b_loc  = (const float*)d_in[5];
  const float* w_mask = (const float*)d_in[6];
  const float* b_mask = (const float*)d_in[7];
  float* out = (float*)d_out;

  char* ws = (char*)d_ws;
  float* y            = (float*)ws;                                  // 134,217,728 B
  unsigned short* xbf = (unsigned short*)(ws + 134217728ull);        //  37,748,736 B
  unsigned short* wBp = (unsigned short*)(ws + 171966464ull);        //   2,654,208 B
  float* gsum         = (float*)(ws + 174620672ull);                 // 512 f
  float* gsumsq       = gsum + 512;
  float* scl          = gsum + 1024;
  float* shf          = gsum + 1536;
  unsigned short* zpd = (unsigned short*)(ws + 174628864ull);        // 256 B zeros

  hipMemsetAsync(gsum, 0, 8448, stream);   // zero stats + zero-pad buffer

  prep_x<<<dim3(BATCH * S), dim3(256), 0, stream>>>(input1, xbf);
  prep_w<<<dim3((9 * CO * CIP + 255) / 256), dim3(256), 0, stream>>>(w_conv, wBp);

  conv_wmma<<<dim3(MTOT / BM, CO / BN), dim3(256), 0, stream>>>(xbf, wBp, zpd, y,
                                                               gsum, gsumsq);

  bn_finalize<<<dim3(2), dim3(256), 0, stream>>>(gsum, gsumsq, gamma, beta, scl, shf);

  decode_kernel<<<dim3(MTOT / 8), dim3(256), 0, stream>>>(y, scl, shf, w_loc, b_loc,
                                                          w_mask, b_mask, out);
}
